// blockSAGEsq_44598940402312
// MI455X (gfx1250) — matrix operand order repaired and checked by reading, not compiled, not run on hardware
//
#include <hip/hip_runtime.h>
#include <hip/hip_bf16.h>
#include <cstddef>

typedef __attribute__((ext_vector_type(16))) _Float16 v16h;
typedef __attribute__((ext_vector_type(8)))  float    v8f;

#define N_NODES 50000
#define N_EDGES 600000
#define HIDDEN  128
#define INNER   256

// ---------------------------------------------------------------------------
// Utility: zero-fill (agg must start at +0.0f so int-atomicMax == segment_max
// over ReLU outputs with isolated-node -> 0 semantics).
// ---------------------------------------------------------------------------
__global__ __launch_bounds__(256) void sage_fill_zero(float* __restrict__ p, size_t n) {
    size_t i = (size_t)blockIdx.x * blockDim.x + threadIdx.x;
    if (i < n) p[i] = 0.0f;
}

// ---------------------------------------------------------------------------
// Scatter-max: for each edge e, agg[dst[e]][:] = max(agg, t[src[e]][:]).
// t >= 0 (post-ReLU), agg zero-initialized, so integer max on the f32 bit
// pattern is exact. One wave (32 lanes) per edge, 4 dims per lane (float4).
// ---------------------------------------------------------------------------
__global__ __launch_bounds__(256) void sage_scatter_max(
    const long long* __restrict__ src, const long long* __restrict__ dst,
    const float* __restrict__ t, int* __restrict__ agg, int nedges) {
    int lane = threadIdx.x & 31;
    int e = blockIdx.x * (blockDim.x >> 5) + (threadIdx.x >> 5);
    if (e >= nedges) return;
    int s = (int)src[e];
    int d = (int)dst[e];
    const float4* ts = (const float4*)(t + (size_t)s * HIDDEN);
    float4 v = ts[lane];                       // dims lane*4 .. lane*4+3
    int* ap = agg + (size_t)d * HIDDEN + lane * 4;
    atomicMax(ap + 0, __float_as_int(v.x));
    atomicMax(ap + 1, __float_as_int(v.y));
    atomicMax(ap + 2, __float_as_int(v.z));
    atomicMax(ap + 3, __float_as_int(v.w));
}

// ---------------------------------------------------------------------------
// WMMA fragment builders (wave32, v_wmma_f32_16x16x32_f16 layouts per ISA).
// ---------------------------------------------------------------------------
// A (16x32 f16): lanes 0-15 hold row M=lane, K = {0..7, 16..23};
//                lanes 16-31 hold row M=lane-16, K = {8..15, 24..31}.
__device__ inline v16h load_a_frag(const float* __restrict__ A, int row, int kc, int lane) {
    const float4* p = (const float4*)(A + (size_t)row * HIDDEN + kc * 32);
    int off = (lane & 16) ? 2 : 0;
    float4 q0 = p[off + 0];   // K +0..3   (or +8..11)
    float4 q1 = p[off + 1];   // K +4..7   (or +12..15)
    float4 q2 = p[off + 4];   // K +16..19 (or +24..27)
    float4 q3 = p[off + 5];   // K +20..23 (or +28..31)
    v16h a;
    a[0]  = (_Float16)q0.x; a[1]  = (_Float16)q0.y; a[2]  = (_Float16)q0.z; a[3]  = (_Float16)q0.w;
    a[4]  = (_Float16)q1.x; a[5]  = (_Float16)q1.y; a[6]  = (_Float16)q1.z; a[7]  = (_Float16)q1.w;
    a[8]  = (_Float16)q2.x; a[9]  = (_Float16)q2.y; a[10] = (_Float16)q2.z; a[11] = (_Float16)q2.w;
    a[12] = (_Float16)q3.x; a[13] = (_Float16)q3.y; a[14] = (_Float16)q3.z; a[15] = (_Float16)q3.w;
    return a;
}

// B (32x16 f16): B[k][n] = W[n][k] (we need A @ W^T). sW is f16 row-major
// [cols][128]. Lane holds column n = lane&15; lane-half selects K {0..7,16..23}
// vs {8..15,24..31} of the 32-chunk; two 8-half chunks -> two ds_load_b128.
__device__ inline v16h load_b_frag(const _Float16* __restrict__ sW, int ncolbase, int kc, int lane) {
    typedef __attribute__((ext_vector_type(8))) _Float16 v8h;
    int col   = ncolbase + (lane & 15);
    int kbase = kc * 32 + ((lane >> 4) & 1) * 8;
    v8h lo = *(const v8h*)(sW + (size_t)col * HIDDEN + kbase);
    v8h hi = *(const v8h*)(sW + (size_t)col * HIDDEN + kbase + 16);
    return __builtin_shufflevector(lo, hi, 0, 1, 2, 3, 4, 5, 6, 7, 8, 9, 10, 11, 12, 13, 14, 15);
}

__device__ inline float gelu_tanh(float x) {
    float x3 = x * x * x;
    float t = tanhf(0.7978845608028654f * (x + 0.044715f * x3));
    return 0.5f * x * (1.0f + t);
}

// ---------------------------------------------------------------------------
// Fused GEMM: out[:, colOff:colOff+128] = act( A0 @ W0^T (+ A1 @ W1^T) + bias )
// A*: [nrows,128] f32.  W*: [>=colOff+128, 128] f32 (slab cached in LDS f16).
// Block = 256 thr = 8 waves; block tile 128 rows; wave tile 16 rows x 128 cols.
// ACT: 0 = ReLU, 1 = tanh-GELU.
// ---------------------------------------------------------------------------
template <bool DUAL, int ACT>
__global__ __launch_bounds__(256) void sage_gemm(
    const float* __restrict__ A0, const float* __restrict__ A1,
    const float* __restrict__ W0, const float* __restrict__ W1,
    const float* __restrict__ bias, float* __restrict__ out,
    int nrows, int ldc) {
    constexpr int NW = DUAL ? 2 : 1;
    __shared__ _Float16 sW[NW * HIDDEN * HIDDEN];   // 32KB (64KB dual)

    const int tid    = threadIdx.x;
    const int lane   = tid & 31;
    const int wave   = tid >> 5;
    const int colOff = blockIdx.y * HIDDEN;          // 0, or 0/128 for final

    // Stage weight slab(s) into LDS as f16.
    for (int i = tid; i < HIDDEN * HIDDEN; i += 256) {
        int r = i >> 7, k = i & 127;
        sW[i] = (_Float16)W0[(size_t)(colOff + r) * HIDDEN + k];
        if (DUAL) sW[HIDDEN * HIDDEN + i] = (_Float16)W1[(size_t)(colOff + r) * HIDDEN + k];
    }
    __syncthreads();

    const int rowbase = blockIdx.x * 128 + wave * 16;
    int arow = rowbase + (lane & 15);
    if (arow > nrows - 1) arow = nrows - 1;          // clamp loads; stores masked

    v8f acc[8];
#pragma unroll
    for (int i = 0; i < 8; ++i)
#pragma unroll
        for (int j = 0; j < 8; ++j) acc[i][j] = 0.0f;

#pragma unroll
    for (int kc = 0; kc < 4; ++kc) {
        v16h a = load_a_frag(A0, arow, kc, lane);
#pragma unroll
        for (int nt = 0; nt < 8; ++nt) {
            v16h b = load_b_frag(sW, nt * 16, kc, lane);
            acc[nt] = __builtin_amdgcn_wmma_f32_16x16x32_f16(
                false, a, false, b, (short)0, acc[nt], false, false);
        }
    }
    if (DUAL) {
#pragma unroll
        for (int kc = 0; kc < 4; ++kc) {
            v16h a = load_a_frag(A1, arow, kc, lane);
#pragma unroll
            for (int nt = 0; nt < 8; ++nt) {
                v16h b = load_b_frag(sW + HIDDEN * HIDDEN, nt * 16, kc, lane);
                acc[nt] = __builtin_amdgcn_wmma_f32_16x16x32_f16(
                    false, a, false, b, (short)0, acc[nt], false, false);
            }
        }
    }

    // Epilogue. C layout: VGPR v -> row rowbase + v + (lane>=16 ? 8 : 0),
    // col = nt*16 + (lane&15).
    const int rsel = (lane & 16) ? 8 : 0;
#pragma unroll
    for (int nt = 0; nt < 8; ++nt) {
        const int gc = colOff + nt * 16 + (lane & 15);
        const float bv = bias[gc];
#pragma unroll
        for (int v = 0; v < 8; ++v) {
            int r = rowbase + rsel + v;
            if (r < nrows) {
                float val = acc[nt][v] + bv;
                val = (ACT == 0) ? fmaxf(val, 0.0f) : gelu_tanh(val);
                out[(size_t)r * ldc + gc] = val;
            }
        }
    }
}

// ---------------------------------------------------------------------------
// Host-side orchestration.
// Inputs: 0:x 1:edge_index 2:Wp1 3:bp1 4:Wl1 5:bl1 6:Wr1
//         7:Wp2 8:bp2 9:Wl2 10:bl2 11:Wr2 12:Wlin 13:blin
// ---------------------------------------------------------------------------
extern "C" void kernel_launch(void* const* d_in, const int* in_sizes, int n_in,
                              void* d_out, int out_size, void* d_ws, size_t ws_size,
                              hipStream_t stream) {
    const float* x    = (const float*)d_in[0];
    const long long* ei = (const long long*)d_in[1];
    const float* Wp1  = (const float*)d_in[2];
    const float* bp1  = (const float*)d_in[3];
    const float* Wl1  = (const float*)d_in[4];
    const float* bl1  = (const float*)d_in[5];
    const float* Wr1  = (const float*)d_in[6];
    const float* Wp2  = (const float*)d_in[7];
    const float* bp2  = (const float*)d_in[8];
    const float* Wl2  = (const float*)d_in[9];
    const float* bl2  = (const float*)d_in[10];
    const float* Wr2  = (const float*)d_in[11];
    const float* Wlin = (const float*)d_in[12];
    const float* blin = (const float*)d_in[13];

    const int N = in_sizes[0] / HIDDEN;    // 50000
    const int E = in_sizes[1] / 2;         // 600000
    const long long* esrc = ei;
    const long long* edst = ei + E;

    const size_t feat = (size_t)N * HIDDEN;
    float* t   = (float*)d_ws;             // transformed features (reused)
    float* agg = t + feat;                 // scatter-max accumulator
    float* h1  = agg + feat;               // layer-1 output
    float* h2  = t;                        // layer-2 output aliases t (t dead)

    const int gb = (N + 127) / 128;                        // row blocks
    const int zb = (int)((feat + 255) / 256);              // fill blocks
    const int sb = (E + 7) / 8;                            // 8 edges / block

    // ---- Layer 1 ----
    sage_fill_zero<<<zb, 256, 0, stream>>>(agg, feat);
    sage_gemm<false, 0><<<gb, 256, 0, stream>>>(x, nullptr, Wp1, nullptr, bp1, t, N, HIDDEN);
    sage_scatter_max<<<sb, 256, 0, stream>>>(esrc, edst, t, (int*)agg, E);
    sage_gemm<true, 1><<<gb, 256, 0, stream>>>(agg, x, Wl1, Wr1, bl1, h1, N, HIDDEN);

    // ---- Layer 2 ----
    sage_fill_zero<<<zb, 256, 0, stream>>>(agg, feat);
    sage_gemm<false, 0><<<gb, 256, 0, stream>>>(h1, nullptr, Wp2, nullptr, bp2, t, N, HIDDEN);
    sage_scatter_max<<<sb, 256, 0, stream>>>(esrc, edst, t, (int*)agg, E);
    sage_gemm<true, 1><<<gb, 256, 0, stream>>>(agg, h1, Wl2, Wr2, bl2, h2, N, HIDDEN);

    // ---- Final linear 128 -> 256, GELU ----
    dim3 gf(gb, INNER / HIDDEN);   // y-dim covers the 256 output columns
    sage_gemm<false, 1><<<gf, 256, 0, stream>>>(h2, nullptr, Wlin, nullptr, blin,
                                                (float*)d_out, N, INNER);

    // ---- Pass-through edge_index (second tuple element) ----
    const size_t hElems = (size_t)N * INNER;
    if ((size_t)out_size > hElems) {
        size_t tailBytes = ((size_t)out_size - hElems) * sizeof(float);
        size_t eiBytes = (size_t)in_sizes[1] * sizeof(long long);
        if (tailBytes > eiBytes) tailBytes = eiBytes;
        hipMemcpyAsync((float*)d_out + hElems, d_in[1], tailBytes,
                       hipMemcpyDeviceToDevice, stream);
    }
}
